// Quantize_65859028517405
// MI455X (gfx1250) — matrix operand order repaired and checked by reading, not compiled, not run on hardware
//
#include <hip/hip_runtime.h>
#include <hip/hip_bf16.h>

// ---------------------------------------------------------------------------
// VQ quantizer forward for MI455X (gfx1250, wave32, WMMA bf16, async->LDS).
//   dims: DIM=512, N_EMBED=8192, N = B*T = 16384
// Pipeline (all on `stream`):
//   0) zero counts + accumulators
//   1) fp32 -> bf16 converts (x, embed, proj_w)
//   2) row norms ||x||^2
//   3) codebook GEMM (WMMA bf16) -> codebook fp32 + bf16
//   4) row norms ||e||^2
//   5) distance GEMM: 64 rows/block, shared double-buffered B tile in LDS
//      filled by GLOBAL_LOAD_ASYNC_TO_LDS_B128 (ASYNCcnt), fused online
//      softmax-entropy + argmax, in-wave shfl butterfly reduction
//   6) gather z_q + MSE reduction
//   7) token-usage entropy + total loss
// ---------------------------------------------------------------------------

#define DIM      512
#define NEMB     8192
#define NTOK     16384                 // B*T = 32*512
#define ZQ_ELEMS ((long)NTOK * (long)DIM)
#define ENT_PEN  0.1f
#define BETA_C   0.25f
#define EPS_V    1e-5f
#define MAX_ENT  9.0109131f            // ln(8192)
#define NEG_BIG  (-3.402823466e38f)

#define BPITCH   520                   // bf16 elems per LDS B row (512 + 8 pad)
#define BTILE    (16 * BPITCH)         // bf16 elems per B tile buffer

typedef __attribute__((ext_vector_type(16))) __bf16 v16bf;
typedef __attribute__((ext_vector_type(8)))  __bf16 v8bf;
typedef __attribute__((ext_vector_type(8)))  float  v8f;

union ABfrag { v16bf v; v8bf h[2]; };

// ---------------------------------------------------------------------------
// Wave-level 16x16 output tile against a row-major global B ([*, DIM] bf16):
// computes A(16xDIM, LDS) x B[brow0..brow0+15]^T, fp32 accumulate.
// A frag (16-bit A layout): lane<16 -> M=lane,    K = {0..7, 16..23}+k0
//                           lane>=16 -> M=lane-16, K = {8..15, 24..31}+k0
// B frag: lane&15 = N (B row), same K split as A frag.
// ---------------------------------------------------------------------------
__device__ inline v8f wmma_rowtile(const __bf16* __restrict__ As,
                                   const __bf16* __restrict__ Bg,
                                   long brow0, int lane)
{
    const int m     = lane & 15;
    const int koff  = (lane >> 4) << 3;   // 0 or 8
    const int khalf = (lane >> 4) << 3;   // 0 or 8
    const __bf16* bptr = Bg + (brow0 + (long)(lane & 15)) * (long)DIM + khalf;
    const __bf16* aptr = As + m * DIM;

    v8f acc = {0.f, 0.f, 0.f, 0.f, 0.f, 0.f, 0.f, 0.f};
#pragma unroll
    for (int k0 = 0; k0 < DIM; k0 += 32) {
        ABfrag a, b;
        a.h[0] = *(const v8bf*)(aptr + k0 + koff);
        a.h[1] = *(const v8bf*)(aptr + k0 + 16 + koff);
        b.h[0] = *(const v8bf*)(bptr + k0);
        b.h[1] = *(const v8bf*)(bptr + k0 + 16);
        acc = __builtin_amdgcn_wmma_f32_16x16x32_bf16(
                  false, a.v, false, b.v, (short)0, acc, false, false);
    }
    return acc;
}

// ---------------------------------------------------------------------------
// small utility kernels
// ---------------------------------------------------------------------------
__global__ void zero_kernel(float* __restrict__ p, long n)
{
    long i = (long)blockIdx.x * blockDim.x + threadIdx.x;
    if (i < n) p[i] = 0.f;
}

__global__ void f32_to_bf16_kernel(const float* __restrict__ src,
                                   __bf16* __restrict__ dst, long n)
{
    long i = (long)blockIdx.x * blockDim.x + threadIdx.x;
    if (i < n) dst[i] = (__bf16)src[i];
}

__global__ __launch_bounds__(128) void rownorm_kernel(const float* __restrict__ X,
                                                      float* __restrict__ nrm,
                                                      int ncols)
{
    const float* row = X + (long)blockIdx.x * ncols;
    float acc = 0.f;
    for (int i = threadIdx.x; i < ncols; i += 128) {
        float v = row[i];
        acc += v * v;
    }
#pragma unroll
    for (int o = 16; o > 0; o >>= 1) acc += __shfl_down(acc, o, 32);
    __shared__ float red[4];
    if ((threadIdx.x & 31) == 0) red[threadIdx.x >> 5] = acc;
    __syncthreads();
    if (threadIdx.x == 0) nrm[blockIdx.x] = red[0] + red[1] + red[2] + red[3];
}

// ---------------------------------------------------------------------------
// codebook = embed @ proj_w^T + b ; emit fp32 (gather/norms) and bf16 (GEMM)
// grid = NEMB/16 blocks, 128 threads (4 waves)
// ---------------------------------------------------------------------------
__global__ __launch_bounds__(128) void codebook_gemm_kernel(
    const __bf16* __restrict__ ebf, const __bf16* __restrict__ wbf,
    const float* __restrict__ bias,
    float* __restrict__ cb_f32, __bf16* __restrict__ cb_bf16)
{
    __shared__ __bf16 As[16 * DIM];
    const int tid  = threadIdx.x;
    const int lane = tid & 31;
    const int wave = tid >> 5;
    const long row0 = (long)blockIdx.x * 16;

    for (int i = tid; i < 16 * DIM / 8; i += 128)
        ((v8bf*)As)[i] = ((const v8bf*)(ebf + row0 * DIM))[i];
    __syncthreads();

    for (int col0 = wave * 16; col0 < DIM; col0 += 64) {
        v8f acc = wmma_rowtile(As, wbf, col0, lane);
        const int n = col0 + (lane & 15);
        const float bv = bias[n];
#pragma unroll
        for (int r = 0; r < 8; ++r) {
            long row = row0 + r + ((lane >> 4) << 3);
            float v = acc[r] + bv;
            cb_f32 [row * DIM + n] = v;
            cb_bf16[row * DIM + n] = (__bf16)v;
        }
    }
}

// ---------------------------------------------------------------------------
// Async fill of one B tile (16 codebook rows x 512 bf16) into LDS buffer,
// via GLOBAL_LOAD_ASYNC_TO_LDS_B128 (per-lane 16B global -> 16B LDS).
// 128 threads x 8 chunks = 1024 x 16B = 16 KB. Tracked by ASYNCcnt.
// ---------------------------------------------------------------------------
__device__ inline void issue_btile_async(const __bf16* __restrict__ cbf,
                                         int col0, unsigned lds_buf_base,
                                         int tid)
{
#pragma unroll
    for (int j = 0; j < 8; ++j) {
        int chunk = tid + j * 128;          // 0..1023
        int row   = chunk >> 6;             // 0..15
        int c8    = chunk & 63;             // 16B chunk within the row
        unsigned      lds = lds_buf_base + (unsigned)(row * (BPITCH * 2) + c8 * 16);
        unsigned long g   = (unsigned long)(const void*)(cbf
                            + (long)(col0 + row) * DIM + c8 * 8);
        asm volatile("global_load_async_to_lds_b128 %0, %1, off"
                     :: "v"(lds), "v"(g) : "memory");
    }
}

// ---------------------------------------------------------------------------
// Main fused kernel: distances + online softmax entropy + argmax.
//   logit(n,k) = 2*dot - ||x_n||^2 - ||e_k||^2
// Block = 128 threads (4 waves), 64 rows: wave w owns rows [row0+16w, +16)
// and sweeps ALL 8192 columns in 16-wide tiles; the B tile is shared by the
// 4 waves from a double-buffered LDS buffer filled asynchronously.
// Online per-lane-slot stats: mx = running max, ss = sum exp(l-mx),
// tt = sum exp(l-mx)*l ; entropy = mx + log ss - tt/ss.
// grid = NTOK/64
// ---------------------------------------------------------------------------
__global__ __launch_bounds__(128) void distance_softmax_kernel(
    const __bf16* __restrict__ xbf,  const float* __restrict__ xnorm,
    const __bf16* __restrict__ cbf,  const float* __restrict__ enorm,
    int*   __restrict__ idx_buf,     float* __restrict__ out_idx_f,
    float* __restrict__ counts,      float* __restrict__ softent_acc)
{
    __shared__ __bf16 Bs[2 * BTILE];     // 2 x 16 x 520 bf16 = 32.5 KB

    const int tid   = threadIdx.x;
    const int lane  = tid & 31;
    const int wave  = tid >> 5;
    const int m     = lane & 15;
    const int koff  = (lane >> 4) << 3;   // 0 or 8  (A frag K offset)
    const int khalf = (lane >> 4) << 3;   // 0 or 8  (B frag K offset)
    const long row0 = (long)blockIdx.x * 64;
    const long wrow0 = row0 + (long)wave * 16;

    const unsigned bs_base = (unsigned)(unsigned long)(const void*)(&Bs[0]);

    // --- A fragments: this wave's 16 rows, register resident (128 VGPRs) ---
    ABfrag af[16];
    {
        const __bf16* xrow = xbf + (wrow0 + m) * (long)DIM;
#pragma unroll
        for (int i = 0; i < 16; ++i) {
            const int k0 = i * 32;
            af[i].h[0] = *(const v8bf*)(xrow + k0 + koff);
            af[i].h[1] = *(const v8bf*)(xrow + k0 + 16 + koff);
        }
    }

    float mx[8], ss[8], tt[8], best[8], xr[8];
    int bidx[8];
#pragma unroll
    for (int r = 0; r < 8; ++r) {
        mx[r] = NEG_BIG; ss[r] = 0.f; tt[r] = 0.f;
        best[r] = NEG_BIG; bidx[r] = 0;
        xr[r] = xnorm[wrow0 + r + ((lane >> 4) << 3)];
    }

    // prologue: async-fill tile 0 into buffer 0
    issue_btile_async(cbf, 0, bs_base, tid);

    const int NT = NEMB / 16;            // 512 column tiles
    for (int t = 0; t < NT; ++t) {
        const int cur = t & 1;

        // tile t resident: wait own async loads, then barrier for all waves'
        asm volatile("s_wait_asynccnt 0x0" ::: "memory");
        __syncthreads();

        if (t + 1 < NT)                  // prefetch next tile into other buffer
            issue_btile_async(cbf, (t + 1) * 16, bs_base + (cur ^ 1) * (BTILE * 2), tid);

        // --- 16 WMMAs on the shared LDS B tile ---
        const __bf16* bp = &Bs[cur * BTILE] + (lane & 15) * BPITCH + khalf;
        v8f acc = {0.f, 0.f, 0.f, 0.f, 0.f, 0.f, 0.f, 0.f};
#pragma unroll
        for (int i = 0; i < 16; ++i) {
            ABfrag b;
            b.h[0] = *(const v8bf*)(bp + i * 32);
            b.h[1] = *(const v8bf*)(bp + i * 32 + 16);
            acc = __builtin_amdgcn_wmma_f32_16x16x32_bf16(
                      false, af[i].v, false, b.v, (short)0, acc, false, false);
        }

        // --- fused epilogue: logits + online softmax/argmax ---
        const int   n  = t * 16 + (lane & 15);
        const float en = enorm[n];
#pragma unroll
        for (int r = 0; r < 8; ++r) {
            float l = 2.0f * acc[r] - xr[r] - en;
            if (l > best[r]) { best[r] = l; bidx[r] = n; }
            if (l > mx[r]) {
                float e = __expf(mx[r] - l);
                ss[r] = ss[r] * e + 1.0f;
                tt[r] = tt[r] * e + l;
                mx[r] = l;
            } else {
                float e = __expf(l - mx[r]);
                ss[r] += e;
                tt[r] += e * l;
            }
        }

        __syncthreads();                 // all waves done reading buf[cur]
    }

    // --- in-wave reduction: butterfly over the 16 lanes of each half-wave ---
#pragma unroll
    for (int o = 1; o < 16; o <<= 1) {
#pragma unroll
        for (int r = 0; r < 8; ++r) {
            float om = __shfl_xor(mx[r],   o, 16);
            float os = __shfl_xor(ss[r],   o, 16);
            float ot = __shfl_xor(tt[r],   o, 16);
            float ob = __shfl_xor(best[r], o, 16);
            int   oi = __shfl_xor(bidx[r], o, 16);
            if (om > mx[r]) {
                float e = __expf(mx[r] - om);
                ss[r] = ss[r] * e + os; tt[r] = tt[r] * e + ot; mx[r] = om;
            } else {
                float e = __expf(om - mx[r]);
                ss[r] += os * e; tt[r] += ot * e;
            }
            if (ob > best[r] || (ob == best[r] && oi < bidx[r])) {
                best[r] = ob; bidx[r] = oi;
            }
        }
    }

    if ((lane & 15) == 0) {              // lanes 0 and 16 finalize 8 rows each
#pragma unroll
        for (int r = 0; r < 8; ++r) {
            long row = wrow0 + ((lane >> 4) << 3) + r;
            float ent = mx[r] + __logf(ss[r]) - tt[r] / ss[r];  // -sum p log p
            atomicAdd(softent_acc, ent);
            idx_buf  [row] = bidx[r];
            out_idx_f[row] = (float)bidx[r];
            atomicAdd(&counts[bidx[r]], 1.0f);
        }
    }
}

// ---------------------------------------------------------------------------
// z_q gather + MSE (diff_loss == commit_loss in value -> one sum suffices)
// ---------------------------------------------------------------------------
__global__ __launch_bounds__(128) void gather_mse_kernel(
    const float* __restrict__ xin, const float* __restrict__ cb,
    const int* __restrict__ idx,
    float* __restrict__ zq, float* __restrict__ mse_acc)
{
    const long row = blockIdx.x;
    const int  k   = idx[row];
    const float* crow = cb  + (long)k * DIM;
    const float* xrow = xin + row * DIM;
    float acc = 0.f;
    for (int i = threadIdx.x; i < DIM; i += 128) {
        float c = crow[i];
        float d = c - xrow[i];
        zq[row * DIM + i] = c;        // z_q_st == z_q numerically
        acc += d * d;
    }
#pragma unroll
    for (int o = 16; o > 0; o >>= 1) acc += __shfl_down(acc, o, 32);
    __shared__ float red[4];
    if ((threadIdx.x & 31) == 0) red[threadIdx.x >> 5] = acc;
    __syncthreads();
    if (threadIdx.x == 0)
        atomicAdd(mse_acc, red[0] + red[1] + red[2] + red[3]);
}

// ---------------------------------------------------------------------------
// token-usage entropy + total loss (1 block, 256 threads)
// accums[0] = sum row soft-entropy, accums[1] = sum (z_q - x)^2
// ---------------------------------------------------------------------------
__global__ __launch_bounds__(256) void finalize_kernel(
    const float* __restrict__ counts, const float* __restrict__ accums,
    float* __restrict__ out_loss)
{
    float acc = 0.f;
    for (int i = threadIdx.x; i < NEMB; i += 256) {
        float a = counts[i] * (1.0f / (float)NTOK) + EPS_V;
        acc += -a * __logf(a);
    }
#pragma unroll
    for (int o = 16; o > 0; o >>= 1) acc += __shfl_down(acc, o, 32);
    __shared__ float red[8];
    if ((threadIdx.x & 31) == 0) red[threadIdx.x >> 5] = acc;
    __syncthreads();
    if (threadIdx.x == 0) {
        float tok_ent = 0.f;
        for (int w = 0; w < 8; ++w) tok_ent += red[w];
        float soft_ent = accums[0] / (float)NTOK;
        float mse      = accums[1] / (float)ZQ_ELEMS;
        float vq  = (1.0f + BETA_C) * mse;
        float sel = ENT_PEN * (1.0f - soft_ent / MAX_ENT);
        float tel = ENT_PEN * (1.0f - tok_ent  / MAX_ENT);
        out_loss[0] = vq + sel + tel;
    }
}

// ---------------------------------------------------------------------------
// workspace layout (bytes, 256-aligned; total ~48.7 MB)
// ---------------------------------------------------------------------------
#define WS_XBF    0L                                    // NTOK*DIM bf16
#define WS_EBF    (WS_XBF   + (long)NTOK * DIM * 2)     // NEMB*DIM bf16
#define WS_WBF    (WS_EBF   + (long)NEMB * DIM * 2)     // DIM*DIM bf16
#define WS_CB32   (WS_WBF   + (long)DIM  * DIM * 2)     // NEMB*DIM f32
#define WS_CBBF   (WS_CB32  + (long)NEMB * DIM * 4)     // NEMB*DIM bf16
#define WS_XNORM  (WS_CBBF  + (long)NEMB * DIM * 2)     // NTOK f32
#define WS_ENORM  (WS_XNORM + (long)NTOK * 4)           // NEMB f32
#define WS_IDX    (WS_ENORM + (long)NEMB * 4)           // NTOK i32
#define WS_COUNTS (WS_IDX   + (long)NTOK * 4)           // NEMB f32
#define WS_ACCUM  (WS_COUNTS+ (long)NEMB * 4)           // [softent_sum, mse_sum]

extern "C" void kernel_launch(void* const* d_in, const int* in_sizes, int n_in,
                              void* d_out, int out_size, void* d_ws, size_t ws_size,
                              hipStream_t stream)
{
    (void)in_sizes; (void)n_in; (void)out_size; (void)ws_size;

    const float* x      = (const float*)d_in[0];   // (32,512,512)
    const float* embed  = (const float*)d_in[1];   // (8192,512)
    const float* proj_w = (const float*)d_in[2];   // (512,512)
    const float* proj_b = (const float*)d_in[3];   // (512,)

    char* ws = (char*)d_ws;
    __bf16* xbf    = (__bf16*)(ws + WS_XBF);
    __bf16* ebf    = (__bf16*)(ws + WS_EBF);
    __bf16* wbf    = (__bf16*)(ws + WS_WBF);
    float*  cb32   = (float*) (ws + WS_CB32);
    __bf16* cbbf   = (__bf16*)(ws + WS_CBBF);
    float*  xnorm  = (float*) (ws + WS_XNORM);
    float*  enorm  = (float*) (ws + WS_ENORM);
    int*    idxbuf = (int*)   (ws + WS_IDX);
    float*  counts = (float*) (ws + WS_COUNTS);
    float*  accums = (float*) (ws + WS_ACCUM);

    float* out_zq   = (float*)d_out;
    float* out_loss = out_zq + ZQ_ELEMS;
    float* out_idx  = out_loss + 1;

    // 0) zero counts (NEMB) + accumulators (contiguous)
    {
        long nz = NEMB + 64;
        zero_kernel<<<dim3((unsigned)((nz + 255) / 256)), dim3(256), 0, stream>>>(counts, nz);
    }
    // 1) fp32 -> bf16
    {
        long n = (long)NTOK * DIM;
        f32_to_bf16_kernel<<<dim3((unsigned)((n + 255) / 256)), dim3(256), 0, stream>>>(x, xbf, n);
        n = (long)NEMB * DIM;
        f32_to_bf16_kernel<<<dim3((unsigned)((n + 255) / 256)), dim3(256), 0, stream>>>(embed, ebf, n);
        n = (long)DIM * DIM;
        f32_to_bf16_kernel<<<dim3((unsigned)((n + 255) / 256)), dim3(256), 0, stream>>>(proj_w, wbf, n);
    }
    // 2) ||x||^2
    rownorm_kernel<<<dim3(NTOK), dim3(128), 0, stream>>>(x, xnorm, DIM);
    // 3) codebook GEMM (WMMA)
    codebook_gemm_kernel<<<dim3(NEMB / 16), dim3(128), 0, stream>>>(ebf, wbf, proj_b, cb32, cbbf);
    // 4) ||e||^2
    rownorm_kernel<<<dim3(NEMB), dim3(128), 0, stream>>>(cb32, enorm, DIM);
    // 5) distances + softmax entropy + argmax (WMMA + async->LDS double buffer)
    distance_softmax_kernel<<<dim3(NTOK / 64), dim3(128), 0, stream>>>(
        xbf, xnorm, cbbf, enorm, idxbuf, out_idx, counts, accums);
    // 6) gather z_q + MSE
    gather_mse_kernel<<<dim3(NTOK), dim3(128), 0, stream>>>(x, cb32, idxbuf, out_zq, accums + 1);
    // 7) total loss
    finalize_kernel<<<dim3(1), dim3(256), 0, stream>>>(counts, accums, out_loss);
}
